// G2ANet_40596030882474
// MI455X (gfx1250) — matrix operand order repaired and checked by reading, not compiled, not run on hardware
//
#include <hip/hip_runtime.h>
#include <hip/hip_bf16.h>
#include <math.h>

// ---------------------------------------------------------------------------
// G2ANet forward for MI455X (gfx1250, wave32, WMMA).
// Heavy GEMMs (edge MLP 263->256->256 over 126976 rows, decoder 512x512)
// run on v_wmma_f32_16x16x32_f16 with f32 accumulation.
// Workspace layout (bytes), total ~9.2 MB:
//   [0)        hm1 tiled f16 (K=288 padded, N=256)      147456
//   [147456)   hm2 tiled f16 (K=256, N=256)             131072
//   [278528)   dec tiled f16 (K=512, N=512)             524288
//   [802816)   agent_embed f32 (4096 x 256)            4194304
//   [4997120)  final_input f16 (4096 x 512)            4194304
// ---------------------------------------------------------------------------

typedef __attribute__((ext_vector_type(16))) _Float16 v16h;
typedef __attribute__((ext_vector_type(8)))  float    v8f;

#define OUT_LOGITS  2097152
#define OUT_UNNORM  2224128
#define OUT_ENT     2351104
#define OUT_HW      2351105
#define OUT_COMB    2478081

__device__ __forceinline__ v8f wmma_f16(v16h a, v16h b, v8f c) {
  return __builtin_amdgcn_wmma_f32_16x16x32_f16(false, a, false, b, (short)0, c,
                                                false, false);
}

// A-fragment k order used here (contiguous, B matches): lane L holds row M=L&15; half h maps to
// K = h + (L>=16 ? 16:0)
__device__ __forceinline__ int kmapA(int lane, int h) {
  return h + ((lane >= 16) ? 16 : 0);
}

__device__ __forceinline__ float wrap_pi(float a) {
  const float PI_F = 3.14159265358979323846f;
  float m = fmodf(a + PI_F, 2.0f * PI_F);
  if (m < 0.f) m += 2.0f * PI_F;
  return m - PI_F;
}

// ---------------------------------------------------------------------------
// Kernel 0: weight preconversion into per-lane WMMA B-fragment order.
// B-fragment order used here (contiguous k, same as A): lane L holds column n=L&15; half h -> K = h + (L>=16?16:0).
// Tiled storage: [((kc*NT + nt)*32 + lane)*16 + h]  -> 32B contiguous per lane.
// ---------------------------------------------------------------------------
__global__ __launch_bounds__(256) void k_prep(
    const float* __restrict__ hm1w, const float* __restrict__ hm2w,
    const float* __restrict__ decw,
    _Float16* __restrict__ hm1t, _Float16* __restrict__ hm2t,
    _Float16* __restrict__ dect, float* __restrict__ entp)
{
  unsigned idx = blockIdx.x * 256u + threadIdx.x;
  if (idx == 0) *entp = 0.f;
  if (idx < 73728u) {                       // hm1: KC=9 (K padded to 288), NT=16
    int h = idx & 15, lane = (idx >> 4) & 31, nt = (idx >> 9) & 15, kc = idx >> 13;
    int k = kc * 32 + h + ((lane >= 16) ? 16 : 0);
    int n = nt * 16 + (lane & 15);
    hm1t[idx] = (k < 263) ? (_Float16)hm1w[n * 263 + k] : (_Float16)0.f;
  } else if (idx < 73728u + 65536u) {       // hm2: KC=8, NT=16
    unsigned i2 = idx - 73728u;
    int h = i2 & 15, lane = (i2 >> 4) & 31, nt = (i2 >> 9) & 15, kc = i2 >> 13;
    int k = kc * 32 + h + ((lane >= 16) ? 16 : 0);
    int n = nt * 16 + (lane & 15);
    hm2t[i2] = (_Float16)hm2w[n * 256 + k];
  } else if (idx < 401408u) {               // dec: KC=16, NT=32
    unsigned i3 = idx - 139264u;
    int h = i3 & 15, lane = (i3 >> 4) & 31, nt = (i3 >> 9) & 31, kc = i3 >> 14;
    int k = kc * 32 + h + ((lane >= 16) ? 16 : 0);
    int n = nt * 16 + (lane & 15);
    dect[i3] = (_Float16)decw[n * 512 + k];
  }
}

// ---------------------------------------------------------------------------
// Kernel 1: agent embedding MLP  (5 -> 128 -> 256, leaky_relu 0.01)
// ---------------------------------------------------------------------------
__global__ __launch_bounds__(128) void k_agent(
    const float* __restrict__ emb, const float* __restrict__ e1w,
    const float* __restrict__ e1b, const float* __restrict__ e2w,
    const float* __restrict__ e2b, float* __restrict__ ae)
{
  const int row = blockIdx.x;
  const int t = threadIdx.x;
  __shared__ float ein[5];
  __shared__ float h1[128];
  if (t < 5) ein[t] = emb[row * 11 + 4 + t];
  __syncthreads();
  float a = e1b[t];
  const float* w = e1w + t * 5;
#pragma unroll
  for (int k = 0; k < 5; ++k) a += ein[k] * w[k];
  h1[t] = a > 0.f ? a : 0.01f * a;
  __syncthreads();
  for (int c = t; c < 256; c += 128) {
    float s = e2b[c];
    const float* w2 = e2w + c * 128;
    for (int k = 0; k < 128; ++k) s += h1[k] * w2[k];
    ae[(size_t)row * 256 + c] = s > 0.f ? s : 0.01f * s;
  }
}

// ---------------------------------------------------------------------------
// Kernel 2: per-(b,i) block. Edge features, hard MLP (WMMA), Gumbel softmax,
// q/k/v, attention, combined weights, entropy, final_input emission.
// 256 threads = 8 waves; each wave owns 2 N-tiles x 2 M-tiles.
// ---------------------------------------------------------------------------
__global__ __launch_bounds__(256) void k_main(
    const float* __restrict__ emb, const float* __restrict__ gum,
    const float* __restrict__ hm1b, const float* __restrict__ hm2b,
    const float* __restrict__ hew, const float* __restrict__ heb,
    const float* __restrict__ qw, const float* __restrict__ kw,
    const float* __restrict__ vw, const float* __restrict__ vb,
    const _Float16* __restrict__ hm1t, const _Float16* __restrict__ hm2t,
    const float* __restrict__ ae, _Float16* __restrict__ fi,
    float* __restrict__ outLog, float* __restrict__ outUn,
    float* __restrict__ outEnt, float* __restrict__ outHw,
    float* __restrict__ outComb)
{
  const int row = blockIdx.x;
  const int b = row >> 5, i = row & 31;
  const int t = threadIdx.x;
  const int lane = t & 31, wv = t >> 5;

  __shared__ float    emb_s[352];       // 32 x 11
  __shared__ float    ae_s[256];
  __shared__ _Float16 aeh[256];
  __shared__ float    se[310];          // soft_edge 31 x 10
  __shared__ _Float16 ef16[256];        // A chunk kc=8: 32 rows x 8 cols
  __shared__ _Float16 H1[32 * 264];     // relu(layer1) f16, padded stride
  __shared__ float    uni[8192];        // union: H2 f32 (32x256) | kOff/vOff f16
  __shared__ float    qs[256];
  __shared__ float    logit_s[64];
  __shared__ float    hw_s[32];
  __shared__ float    sc_s[32];
  __shared__ float    comb_s[32];

  float*    H2   = uni;
  _Float16* kOff = (_Float16*)uni;             // 31 x 256 f16
  _Float16* vOff = (_Float16*)(uni + 4096);    // 31 x 256 f16

  // ---- load embedding tile + agent embed row -----------------------------
  if (t < 32) {
    for (int f = 0; f < 11; ++f) emb_s[t * 11 + f] = emb[(b * 32 + t) * 11 + f];
  }
  ae_s[t] = ae[(size_t)row * 256 + t];
  aeh[t]  = (_Float16)ae_s[t];
  __syncthreads();

  // ---- edge + goal-polar features (thread t = neighbor j) ----------------
  if (t < 32) {
    if (t != i) {
      int jj = t - (t > i ? 1 : 0);
      const float* Ei = emb_s + i * 11;
      const float* Ej = emb_s + t * 11;
      float dx = Ej[0] - Ei[0], dy = Ej[1] - Ei[1];
      float un = sqrtf(dx * dx + dy * dy);
      float hi = atan2f(Ei[3], Ei[2]);
      float ang = wrap_pi(atan2f(dy, dx) - hi);
      float* s = se + jj * 10;
      s[0] = un * (1.f / 12.f);
      s[1] = cosf(ang);
      s[2] = sinf(ang);
      s[3] = Ej[2]; s[4] = Ej[3];
      s[5] = Ej[7]; s[6] = Ej[8];
      float gx = Ej[9] - Ei[0], gy = Ej[10] - Ei[1];
      float ga = wrap_pi(atan2f(gy, gx) - hi);
      s[7] = sqrtf(gx * gx + gy * gy);
      s[8] = cosf(ga);
      s[9] = sinf(ga);
      outUn[row * 31 + jj] = un;
      for (int f = 0; f < 7; ++f) ef16[jj * 8 + f] = (_Float16)s[f];
      ef16[jj * 8 + 7] = (_Float16)0.f;
    } else {
      for (int f = 0; f < 8; ++f) ef16[31 * 8 + f] = (_Float16)0.f;  // pad row 31
    }
  }
  // ---- q projection (VALU, q_w stays L2-resident) ------------------------
  {
    float qv = 0.f;
    const float* qr = qw + t * 256;
    for (int k2 = 0; k2 < 256; ++k2) qv += ae_s[k2] * qr[k2];
    qs[t] = qv;
  }
  __syncthreads();

  // ---- hard MLP layer 1: (32 rows x K=288) @ (288 x 256), WMMA -----------
  // hard_input cols 0..255 == agent_embed[b,i] for every row -> A fragment
  // is row-independent for kc<8; only kc=8 (edge features) is per-row.
  v8f acc[2][2] = {};
  for (int kc = 0; kc < 9; ++kc) {
    v16h af0, af1;
    if (kc < 8) {
#pragma unroll
      for (int h = 0; h < 16; ++h) {
        int kl = kmapA(lane, h);
        af0[h] = aeh[kc * 32 + kl];
      }
      af1 = af0;
    } else {
#pragma unroll
      for (int h = 0; h < 16; ++h) {
        int kl = kmapA(lane, h);
        int m0 = lane & 15;
        af0[h] = (kl < 7) ? ef16[m0 * 8 + kl] : (_Float16)0.f;
        af1[h] = (kl < 7) ? ef16[(m0 + 16) * 8 + kl] : (_Float16)0.f;
      }
    }
#pragma unroll
    for (int u = 0; u < 2; ++u) {
      int nt = wv * 2 + u;
      v16h bf = *(const v16h*)(hm1t + (((kc * 16 + nt) * 32 + lane) << 4));
      acc[0][u] = wmma_f16(af0, bf, acc[0][u]);
      acc[1][u] = wmma_f16(af1, bf, acc[1][u]);
    }
  }
  // bias + relu -> H1 (f16)
#pragma unroll
  for (int mt = 0; mt < 2; ++mt)
#pragma unroll
    for (int u = 0; u < 2; ++u) {
      int n = (wv * 2 + u) * 16 + (lane & 15);
      float bia = hm1b[n];
#pragma unroll
      for (int e = 0; e < 8; ++e) {
        int m = mt * 16 + e + ((lane >= 16) ? 8 : 0);
        float vv = acc[mt][u][e] + bia;
        H1[m * 264 + n] = (_Float16)(vv > 0.f ? vv : 0.f);
      }
    }
  __syncthreads();

  // ---- hard MLP layer 2: (32 x 256) @ (256 x 256), WMMA ------------------
  v8f acc2[2][2] = {};
  for (int kc = 0; kc < 8; ++kc) {
    v16h af0, af1;
#pragma unroll
    for (int h = 0; h < 16; ++h) {
      int kl = kmapA(lane, h);
      int m0 = lane & 15;
      af0[h] = H1[m0 * 264 + kc * 32 + kl];
      af1[h] = H1[(m0 + 16) * 264 + kc * 32 + kl];
    }
#pragma unroll
    for (int u = 0; u < 2; ++u) {
      int nt = wv * 2 + u;
      v16h bf = *(const v16h*)(hm2t + (((kc * 16 + nt) * 32 + lane) << 4));
      acc2[0][u] = wmma_f16(af0, bf, acc2[0][u]);
      acc2[1][u] = wmma_f16(af1, bf, acc2[1][u]);
    }
  }
#pragma unroll
  for (int mt = 0; mt < 2; ++mt)
#pragma unroll
    for (int u = 0; u < 2; ++u) {
      int n = (wv * 2 + u) * 16 + (lane & 15);
      float bia = hm2b[n];
#pragma unroll
      for (int e = 0; e < 8; ++e) {
        int m = mt * 16 + e + ((lane >= 16) ? 8 : 0);
        H2[m * 256 + n] = acc2[mt][u][e] + bia;
      }
    }
  __syncthreads();

  // ---- hard logits + Gumbel-softmax --------------------------------------
  if (t < 62) {
    int r = t >> 1, c = t & 1;
    const float* hwp = hew + c * 256;
    float s = heb[c];
    for (int n = 0; n < 256; ++n) s += H2[r * 256 + n] * hwp[n];
    logit_s[r * 2 + c] = s;
    if (c == 1) outLog[row * 31 + r] = s;
  }
  __syncthreads();
  if (t < 31) {
    float l0 = logit_s[t * 2 + 0], l1 = logit_s[t * 2 + 1];
    float u0 = gum[((size_t)row * 31 + t) * 2 + 0];
    float u1 = gum[((size_t)row * 31 + t) * 2 + 1];
    float g0 = -logf(-logf(u0 + 1e-10f) + 1e-10f);
    float g1 = -logf(-logf(u1 + 1e-10f) + 1e-10f);
    float y0 = (l0 + g0) * 2.0f;   // / TAU, TAU = 0.5
    float y1 = (l1 + g1) * 2.0f;
    float p1 = 1.f / (1.f + expf(y0 - y1));
    hw_s[t] = p1;
    outHw[row * 31 + t] = p1;
  }
  __syncthreads();   // H2 reads done; union region now reused for k/v

  // ---- k/v projections (K=10, VALU) --------------------------------------
  {
    float kreg[10], vreg[10];
#pragma unroll
    for (int f = 0; f < 10; ++f) {
      kreg[f] = kw[t * 10 + f];
      vreg[f] = vw[t * 10 + f];
    }
    float vbi = vb[t];
    for (int jj = 0; jj < 31; ++jj) {
      const float* s = se + jj * 10;
      float kk = 0.f, vv = vbi;
#pragma unroll
      for (int f = 0; f < 10; ++f) { kk += s[f] * kreg[f]; vv += s[f] * vreg[f]; }
      kOff[jj * 256 + t] = (_Float16)kk;
      vOff[jj * 256 + t] = (_Float16)(vv > 0.f ? vv : 0.f);
    }
  }
  __syncthreads();

  // ---- attention scores (8 threads per neighbor, shfl reduce) ------------
  {
    int j = t >> 3, sub = t & 7;
    float part = 0.f;
    if (j < 31)
      for (int c = sub; c < 256; c += 8) part += qs[c] * (float)kOff[j * 256 + c];
    part += __shfl_xor(part, 1, 32);
    part += __shfl_xor(part, 2, 32);
    part += __shfl_xor(part, 4, 32);
    if (j < 31 && sub == 0) sc_s[j] = part * 0.0625f;   // / sqrt(256)
  }
  __syncthreads();

  // ---- softmax, combined weights, entropy (31 elems, serial is cheap) ----
  if (t == 0) {
    float mx = -1e30f;
    for (int j = 0; j < 31; ++j) mx = fmaxf(mx, sc_s[j]);
    float sum = 0.f;
    for (int j = 0; j < 31; ++j) { float e = expf(sc_s[j] - mx); sc_s[j] = e; sum += e; }
    float inv = 1.f / sum;
    float csum = 0.f;
    for (int j = 0; j < 31; ++j) {
      float cmb = sc_s[j] * inv * hw_s[j];
      comb_s[j] = cmb;
      csum += cmb;
    }
    float den = csum + 1e-6f;
    float ent = 0.f;
    for (int j = 0; j < 31; ++j) {
      float cwv = comb_s[j] / den;
      ent -= cwv * logf(cwv + 1e-6f);
    }
    atomicAdd(outEnt, ent * (1.0f / 4096.0f));
  }
  __syncthreads();
  if (t < 31) outComb[row * 31 + t] = comb_s[t];

  // ---- weighted value sum + final_input (f16) ----------------------------
  {
    float xv = 0.f;
    for (int j = 0; j < 31; ++j) xv += (float)vOff[j * 256 + t] * comb_s[j];
    fi[(size_t)row * 512 + t]       = aeh[t];
    fi[(size_t)row * 512 + 256 + t] = (_Float16)xv;
  }
}

// ---------------------------------------------------------------------------
// Kernel 3: decoder  (4096 x 512) @ (512 x 512), WMMA, 16-row M tiles.
// ---------------------------------------------------------------------------
__global__ __launch_bounds__(256) void k_dec(
    const _Float16* __restrict__ fi, const _Float16* __restrict__ dect,
    const float* __restrict__ decb, float* __restrict__ out)
{
  __shared__ _Float16 As[16 * 512];
  const int t = threadIdx.x, lane = t & 31, wv = t >> 5;
  const int r0 = blockIdx.x << 4;

  const uint4* src = (const uint4*)(fi + (size_t)r0 * 512);
  uint4* dst = (uint4*)As;
  for (int idx = t; idx < 1024; idx += 256) dst[idx] = src[idx];
  __syncthreads();

  v8f acc[4] = {};
  for (int kc = 0; kc < 16; ++kc) {
    v16h a;
#pragma unroll
    for (int h = 0; h < 16; ++h) {
      int kl = kmapA(lane, h);
      a[h] = As[(lane & 15) * 512 + kc * 32 + kl];
    }
#pragma unroll
    for (int u = 0; u < 4; ++u) {
      int nt = wv * 4 + u;
      v16h bf = *(const v16h*)(dect + (((kc * 32 + nt) * 32 + lane) << 4));
      acc[u] = wmma_f16(a, bf, acc[u]);
    }
  }
#pragma unroll
  for (int u = 0; u < 4; ++u) {
    int n = (wv * 4 + u) * 16 + (lane & 15);
    float bb = decb[n];
#pragma unroll
    for (int e = 0; e < 8; ++e) {
      int m = e + ((lane >= 16) ? 8 : 0);
      out[(size_t)(r0 + m) * 512 + n] = acc[u][e] + bb;
    }
  }
}

// ---------------------------------------------------------------------------
extern "C" void kernel_launch(void* const* d_in, const int* in_sizes, int n_in,
                              void* d_out, int out_size, void* d_ws, size_t ws_size,
                              hipStream_t stream) {
  const float* embedding = (const float*)d_in[0];
  const float* gumbel    = (const float*)d_in[1];
  const float* e1w       = (const float*)d_in[2];
  const float* e1b       = (const float*)d_in[3];
  const float* e2w       = (const float*)d_in[4];
  const float* e2b       = (const float*)d_in[5];
  const float* hm1w      = (const float*)d_in[6];
  const float* hm1b      = (const float*)d_in[7];
  const float* hm2w      = (const float*)d_in[8];
  const float* hm2b      = (const float*)d_in[9];
  const float* hew       = (const float*)d_in[10];
  const float* heb       = (const float*)d_in[11];
  const float* qw        = (const float*)d_in[12];
  const float* kwp       = (const float*)d_in[13];
  const float* vwp       = (const float*)d_in[14];
  const float* vbp       = (const float*)d_in[15];
  const float* decw      = (const float*)d_in[16];
  const float* decb      = (const float*)d_in[17];

  float* out = (float*)d_out;
  char*  ws  = (char*)d_ws;
  _Float16* hm1t = (_Float16*)(ws + 0);
  _Float16* hm2t = (_Float16*)(ws + 147456);
  _Float16* dect = (_Float16*)(ws + 278528);
  float*    ae   = (float*)(ws + 802816);
  _Float16* fi   = (_Float16*)(ws + 4997120);

  k_prep<<<1568, 256, 0, stream>>>(hm1w, hm2w, decw, hm1t, hm2t, dect,
                                   out + OUT_ENT);
  k_agent<<<4096, 128, 0, stream>>>(embedding, e1w, e1b, e2w, e2b, ae);
  k_main<<<4096, 256, 0, stream>>>(embedding, gumbel, hm1b, hm2b, hew, heb,
                                   qw, kwp, vwp, vbp, hm1t, hm2t, ae, fi,
                                   out + OUT_LOGITS, out + OUT_UNNORM,
                                   out + OUT_ENT, out + OUT_HW, out + OUT_COMB);
  k_dec<<<256, 256, 0, stream>>>(fi, dect, decb, out);
}
